// Bidirectional_LSTM_Cell_Decoder_41360535060842
// MI455X (gfx1250) — matrix operand order repaired and checked by reading, not compiled, not run on hardware
//
#include <hip/hip_runtime.h>
#include <hip/hip_bf16.h>
#include <math.h>

// ---------------------------------------------------------------------------
// Bidirectional 2-layer LSTM greedy decoder for gfx1250 (MI455X).
//  * GEMMs:   v_wmma_f32_16x16x32_bf16 (bf16 A/B, f32 accumulate)
//  * Weights: converted to bf16 once into d_ws (~68MB -> L2-resident in 192MB)
//  * B-tiles: staged per-block into LDS via gfx1250 async-to-LDS DMA
//             (global_load_async_to_lds_b128 + s_wait_asynccnt), double
//             buffered so the next slab's DMA overlaps this slab's WMMAs and
//             the 8 waves of a block read weights from LDS instead of each
//             refetching them through the WGP$.
// ---------------------------------------------------------------------------

#define BB   128
#define TT   256
#define HH   1024
#define EE   1024
#define LL   2
#define EOS_ID 1

typedef __attribute__((ext_vector_type(16))) __bf16          v16bf;
typedef __attribute__((ext_vector_type(8)))  float           v8f;
typedef __attribute__((ext_vector_type(8)))  unsigned short  v8u;
typedef __attribute__((ext_vector_type(16))) unsigned short  v16u;

union BfPack16 { v16u u; v16bf b; };
union BfSplit  { v16u v; v8u h[2]; };

static __device__ __forceinline__ v16bf load_bf16_contig(const unsigned short* p) {
    BfPack16 t;
    t.u = *(const v16u*)p;
    return t.b;
}

// A-matrix per-lane k order used here (same as B): vector elems 0..7 -> K = base..base+7,
// elems 8..15 -> K = base+8..base+15 (base = lane<16 ? 0 : 16).
static __device__ __forceinline__ v16bf load_bf16_split(const unsigned short* p_lo,
                                                        const unsigned short* p_hi) {
    BfSplit s;
    s.h[0] = *(const v8u*)p_lo;
    s.h[1] = *(const v8u*)p_hi;
    BfPack16 t;
    t.u = s.v;
    return t.b;
}

static __device__ __forceinline__ unsigned short f32_to_bf16_rne(float f) {
    unsigned int u = __float_as_uint(f);
    unsigned int r = (u + 0x7FFFu + ((u >> 16) & 1u)) >> 16;
    return (unsigned short)r;
}

static __device__ __forceinline__ float sigmoidf_fast(float x) {
    return 1.0f / (1.0f + __expf(-x));
}

// ---------------------------------------------------------------------------
// One LSTM cell step for one (layer, direction):
//   z = x @ Wih^T + h @ Whh^T + bih + bhh ;  i,f,g,o = split(z)
//   c' = sig(f)*c + sig(i)*tanh(g) ; h' = sig(o)*tanh(c')
//
// grid.x = HH/16 column chunks; block = 256 threads = 8 waves, wave w owns
// batch rows [16w,16w+16).  K is consumed in 64 slabs of 32 (32 slabs of the
// x@Wih^T part, then 32 slabs of h@Whh^T).  Each slab's 4 gate B-tiles
// (4KB) are DMA'd into LDS with async-to-LDS loads, double buffered.
// ---------------------------------------------------------------------------
__global__ __launch_bounds__(256)
void lstm_cell_kernel(const unsigned short* __restrict__ xin,   // [128][1024] bf16
                      const unsigned short* __restrict__ hin,   // [128][1024] bf16
                      float* __restrict__ c,                    // [128][1024] f32 (in/out)
                      const unsigned short* __restrict__ Wih,   // [4096][1024] bf16
                      const unsigned short* __restrict__ Whh,   // [4096][1024] bf16
                      const float* __restrict__ bih,            // [4096]
                      const float* __restrict__ bhh,            // [4096]
                      unsigned short* __restrict__ hout)        // [128][1024] bf16
{
    // [buffer][gate][row(16)][k(32)] : 2 x 4KB
    __shared__ __align__(32) unsigned short smem[2][4][16][32];

    const int tid    = threadIdx.x;
    const int lane   = tid & 31;
    const int wave   = tid >> 5;
    const int n0     = blockIdx.x * 16;
    const int m0     = wave * 16;
    const int nlane  = n0 + (lane & 15);
    const int aShift = (lane < 16) ? 0 : 16;
    const int bShift = (lane < 16) ? 0 : 16;
    const int arow   = m0 + (lane & 15);

    // --- staging map: 256 threads x 16B = one 4KB slab per issue ---
    const int sg = tid >> 6;          // gate 0..3
    const int sr = (tid >> 2) & 15;   // row within tile
    const int sp = tid & 3;           // 16-byte part of the 64B row
    const unsigned ldsBase =
        (unsigned)(unsigned long long)(uintptr_t)&smem[0][0][0][0];
    const unsigned myLds = ldsBase + (unsigned)(sg * 1024 + sr * 64 + sp * 16);
    const size_t   stgRow = (size_t)(sg * HH + n0 + sr) * HH;

    auto issue_slab = [&](int slab) {
        const unsigned short* Wm = (slab < 32) ? Wih : Whh;
        const int kb = (slab & 31) * 32;
        const unsigned short* ga = Wm + stgRow + kb + sp * 8;
        const unsigned dst = myLds + (unsigned)((slab & 1) * 4096);
        asm volatile("global_load_async_to_lds_b128 %0, %1, off"
                     :: "v"(dst), "v"(ga)
                     : "memory");
    };

    v8f acc[4] = {v8f{}, v8f{}, v8f{}, v8f{}};

    issue_slab(0);
    for (int i = 0; i < 64; ++i) {
        if (i + 1 < 64) {
            issue_slab(i + 1);
            asm volatile("s_wait_asynccnt 0x1" ::: "memory"); // slab i landed
        } else {
            asm volatile("s_wait_asynccnt 0x0" ::: "memory");
        }
        __syncthreads();   // all waves' DMA portions visible

        // A tile (x for slabs 0..31, h for slabs 32..63) straight from global.
        const unsigned short* Arow =
            ((i < 32) ? xin : hin) + (size_t)arow * HH;
        const int kb = (i & 31) * 32;
        v16bf a = load_bf16_split(Arow + kb + aShift, Arow + kb + aShift + 8);

        // B tiles from LDS (ds_load_b128 pairs).
        const unsigned short* sb = &smem[i & 1][0][lane & 15][bShift];
#pragma unroll
        for (int g = 0; g < 4; ++g) {
            v16bf b = load_bf16_contig(sb + g * 512);   // gate stride = 16*32 halves
            acc[g] = __builtin_amdgcn_wmma_f32_16x16x32_bf16(
                false, a, false, b, (short)0, acc[g], false, false);
        }
        __syncthreads();   // everyone done reading before buffer is overwritten
    }

    // ---- elementwise cell update (each lane owns 8 (m,n) elements) ----
    const float bi_i = bih[0 * HH + nlane] + bhh[0 * HH + nlane];
    const float bi_f = bih[1 * HH + nlane] + bhh[1 * HH + nlane];
    const float bi_g = bih[2 * HH + nlane] + bhh[2 * HH + nlane];
    const float bi_o = bih[3 * HH + nlane] + bhh[3 * HH + nlane];

#pragma unroll
    for (int r = 0; r < 8; ++r) {
        const int m = m0 + ((lane < 16) ? r : r + 8);
        const size_t idx = (size_t)m * HH + nlane;
        const float zi = acc[0][r] + bi_i;
        const float zf = acc[1][r] + bi_f;
        const float zg = acc[2][r] + bi_g;
        const float zo = acc[3][r] + bi_o;
        const float cn = sigmoidf_fast(zf) * c[idx] + sigmoidf_fast(zi) * tanhf(zg);
        const float hn = sigmoidf_fast(zo) * tanhf(cn);
        c[idx]    = cn;
        hout[idx] = f32_to_bf16_rne(hn);
    }
}

// ---------------------------------------------------------------------------
// logits = wh @ fc_W^T + fc_b   (M=128, N=1024, K=2048) via WMMA bf16.
// ---------------------------------------------------------------------------
__global__ __launch_bounds__(256)
void fc_gemm_kernel(const unsigned short* __restrict__ wh,     // [128][2048] bf16
                    const unsigned short* __restrict__ fcW,    // [1024][2048] bf16
                    const float* __restrict__ fcb,             // [1024]
                    float* __restrict__ logits)                // [128][1024]
{
    const int lane   = threadIdx.x & 31;
    const int wave   = threadIdx.x >> 5;
    const int n0     = blockIdx.x * 16;
    const int m0     = wave * 16;
    const int nlane  = n0 + (lane & 15);
    const int aShift = (lane < 16) ? 0 : 16;
    const int bShift = (lane < 16) ? 0 : 16;
    const int arow   = m0 + (lane & 15);
    const int K      = 2 * HH;

    v8f acc = {};
    const unsigned short* Arow = wh  + (size_t)arow  * K;
    const unsigned short* Brow = fcW + (size_t)nlane * K;

    for (int kb = 0; kb < K; kb += 32) {
        v16bf a = load_bf16_split(Arow + kb + aShift, Arow + kb + aShift + 8);
        v16bf b = load_bf16_contig(Brow + kb + bShift);
        acc = __builtin_amdgcn_wmma_f32_16x16x32_bf16(
            false, a, false, b, (short)0, acc, false, false);
    }

    const float bias = fcb[nlane];
#pragma unroll
    for (int r = 0; r < 8; ++r) {
        const int m = m0 + ((lane < 16) ? r : r + 8);
        logits[(size_t)m * EE + nlane] = acc[r] + bias;
    }
}

// ---------------------------------------------------------------------------
// wh = concat(h_fwd, h_rev).T.reshape(B, 2H) -> wh[b][j] = concat[j%128][b*16+j/128]
// ---------------------------------------------------------------------------
__global__ void repack_wh_kernel(const unsigned short* __restrict__ hf,
                                 const unsigned short* __restrict__ hr,
                                 unsigned short* __restrict__ wh)
{
    int i = blockIdx.x * blockDim.x + threadIdx.x;
    if (i >= BB * 2 * HH) return;
    const int b = i >> 11;
    const int j = i & 2047;
    const int r = j & 127;
    const int col = b * 16 + (j >> 7);
    wh[i] = (col < HH) ? hf[(size_t)r * HH + col]
                       : hr[(size_t)r * HH + (col - HH)];
}

// ---------------------------------------------------------------------------
// Per-batch-row: ended substitution, argmax, ended update, softmax -> d_out,
// and gather next-step embedding (bf16).
// ---------------------------------------------------------------------------
__global__ __launch_bounds__(256)
void decide_kernel(const float* __restrict__ logits,          // [128][1024]
                   int* __restrict__ ended,                   // [128]
                   const unsigned short* __restrict__ emb_bf, // [1024][1024]
                   unsigned short* __restrict__ xout,         // [128][1024]
                   float* __restrict__ out,                   // [128][256][1024]
                   int t)
{
    const int b   = blockIdx.x;
    const int tid = threadIdx.x;
    const bool isEnded = (ended[b] != 0);
    const float* row = logits + (size_t)b * EE;

    __shared__ float sv[256];
    __shared__ int   si[256];
    __shared__ float ss[256];

    float best = -INFINITY;
    int   bidx = EE;
    for (int j = tid; j < EE; j += 256) {
        float v = isEnded ? ((j == EOS_ID) ? 1.0f : 0.0f) : row[j];
        if (v > best || (v == best && j < bidx)) { best = v; bidx = j; }
    }
    sv[tid] = best; si[tid] = bidx;
    __syncthreads();
    for (int s = 128; s > 0; s >>= 1) {
        if (tid < s) {
            if (sv[tid + s] > sv[tid] ||
                (sv[tid + s] == sv[tid] && si[tid + s] < si[tid])) {
                sv[tid] = sv[tid + s];
                si[tid] = si[tid + s];
            }
        }
        __syncthreads();
    }
    const int   label = si[0];
    const float maxv  = sv[0];

    float lsum = 0.0f;
    for (int j = tid; j < EE; j += 256) {
        float v = isEnded ? ((j == EOS_ID) ? 1.0f : 0.0f) : row[j];
        lsum += __expf(v - maxv);
    }
    ss[tid] = lsum;
    __syncthreads();
    for (int s = 128; s > 0; s >>= 1) {
        if (tid < s) ss[tid] += ss[tid + s];
        __syncthreads();
    }
    const float inv = 1.0f / ss[0];

    float* orow = out + ((size_t)b * TT + t) * EE;
    const unsigned short* erow = emb_bf + (size_t)label * EE;
    for (int j = tid; j < EE; j += 256) {
        float v = isEnded ? ((j == EOS_ID) ? 1.0f : 0.0f) : row[j];
        orow[j] = __expf(v - maxv) * inv;
        xout[(size_t)b * EE + j] = erow[j];
    }
    if (tid == 0) ended[b] = (isEnded || label == EOS_ID) ? 1 : 0;
}

// ---------------------------------------------------------------------------
// Init / conversion helpers.
// ---------------------------------------------------------------------------
__global__ void cvt_f32_bf16_kernel(const float* __restrict__ src,
                                    unsigned short* __restrict__ dst, int n)
{
    for (int i = blockIdx.x * blockDim.x + threadIdx.x; i < n;
         i += gridDim.x * blockDim.x)
        dst[i] = f32_to_bf16_rne(src[i]);
}

// h_state layout: [layer][dir][pingpong][128*1024] bf16
__global__ void init_h_kernel(const float* __restrict__ h_t,
                              const float* __restrict__ h_t_rev,
                              unsigned short* __restrict__ hs)
{
    const int n = LL * BB * HH;
    for (int i = blockIdx.x * blockDim.x + threadIdx.x; i < n;
         i += gridDim.x * blockDim.x) {
        const int l = i >> 17;
        const int rest = i & 131071;
        hs[(size_t)(((l * 2 + 0) * 2 + 0)) * (BB * HH) + rest] = f32_to_bf16_rne(h_t[i]);
        hs[(size_t)(((l * 2 + 1) * 2 + 0)) * (BB * HH) + rest] = f32_to_bf16_rne(h_t_rev[i]);
    }
}

// c_state layout: [layer][dir][128*1024] f32, broadcast of c0[l][:]
__global__ void init_c_kernel(const float* __restrict__ c0,
                              const float* __restrict__ c0_rev,
                              float* __restrict__ cs)
{
    const int n = LL * BB * HH;
    for (int i = blockIdx.x * blockDim.x + threadIdx.x; i < n;
         i += gridDim.x * blockDim.x) {
        const int l = i >> 17;
        const int rest = i & 131071;
        const int col = rest & (HH - 1);
        cs[(size_t)(l * 2 + 0) * (BB * HH) + rest] = c0[l * HH + col];
        cs[(size_t)(l * 2 + 1) * (BB * HH) + rest] = c0_rev[l * HH + col];
    }
}

__global__ void init_ended_kernel(int* __restrict__ ended)
{
    if (threadIdx.x < BB) ended[threadIdx.x] = 0;
}

__global__ void gather_first_kernel(const int* __restrict__ yy_pad,
                                    const unsigned short* __restrict__ emb_bf,
                                    unsigned short* __restrict__ xout)
{
    const int b = blockIdx.x;
    const int feed = yy_pad[(size_t)b * TT];
    const unsigned short* erow = emb_bf + (size_t)feed * EE;
    for (int j = threadIdx.x; j < EE; j += blockDim.x)
        xout[(size_t)b * EE + j] = erow[j];
}

// ---------------------------------------------------------------------------
// Host driver: one-time conversion + 256-step decode loop (graph-capturable).
// ---------------------------------------------------------------------------
extern "C" void kernel_launch(void* const* d_in, const int* in_sizes, int n_in,
                              void* d_out, int out_size, void* d_ws, size_t ws_size,
                              hipStream_t stream)
{
    const int*   yy_pad   = (const int*)  d_in[0];
    const float* h_t      = (const float*)d_in[1];
    const float* h_t_rev  = (const float*)d_in[2];
    /* d_in[3] = x_lens, unused by the reference */
    const float* emb      = (const float*)d_in[4];
    const float* W_ih     = (const float*)d_in[5];
    const float* W_hh     = (const float*)d_in[6];
    const float* b_ih     = (const float*)d_in[7];
    const float* b_hh     = (const float*)d_in[8];
    const float* W_ih_r   = (const float*)d_in[9];
    const float* W_hh_r   = (const float*)d_in[10];
    const float* b_ih_r   = (const float*)d_in[11];
    const float* b_hh_r   = (const float*)d_in[12];
    const float* c0       = (const float*)d_in[13];
    const float* c0_rev   = (const float*)d_in[14];
    const float* fc_W     = (const float*)d_in[15];
    const float* fc_b     = (const float*)d_in[16];
    float* out = (float*)d_out;

    // ---- carve workspace ----
    char* ws = (char*)d_ws;
    size_t off = 0;
    auto carve = [&](size_t bytes) -> void* {
        void* p = ws + off;
        off += (bytes + 255) & ~(size_t)255;
        return p;
    };
    const size_t wsz = (size_t)LL * 4 * HH * HH;   // elements per big weight tensor
    unsigned short* wih_bf  = (unsigned short*)carve(wsz * 2);
    unsigned short* whh_bf  = (unsigned short*)carve(wsz * 2);
    unsigned short* wihr_bf = (unsigned short*)carve(wsz * 2);
    unsigned short* whhr_bf = (unsigned short*)carve(wsz * 2);
    unsigned short* fc_bf   = (unsigned short*)carve((size_t)EE * 2 * HH * 2);
    unsigned short* emb_bf  = (unsigned short*)carve((size_t)EE * EE * 2);
    unsigned short* h_state = (unsigned short*)carve((size_t)LL * 2 * 2 * BB * HH * 2);
    float*          c_state = (float*)         carve((size_t)LL * 2 * BB * HH * 4);
    unsigned short* x_bf    = (unsigned short*)carve((size_t)BB * HH * 2);
    unsigned short* wh_bf   = (unsigned short*)carve((size_t)BB * 2 * HH * 2);
    float*          logits  = (float*)         carve((size_t)BB * EE * 4);
    int*            ended   = (int*)           carve((size_t)BB * 4);

    auto hbuf = [&](int l, int d, int pp) -> unsigned short* {
        return h_state + (size_t)(((l * 2 + d) * 2 + pp)) * (BB * HH);
    };
    auto cbuf = [&](int l, int d) -> float* {
        return c_state + (size_t)(l * 2 + d) * (BB * HH);
    };

    // ---- one-time conversions & state init ----
    cvt_f32_bf16_kernel<<<2048, 256, 0, stream>>>(W_ih,   wih_bf,  (int)wsz);
    cvt_f32_bf16_kernel<<<2048, 256, 0, stream>>>(W_hh,   whh_bf,  (int)wsz);
    cvt_f32_bf16_kernel<<<2048, 256, 0, stream>>>(W_ih_r, wihr_bf, (int)wsz);
    cvt_f32_bf16_kernel<<<2048, 256, 0, stream>>>(W_hh_r, whhr_bf, (int)wsz);
    cvt_f32_bf16_kernel<<<1024, 256, 0, stream>>>(fc_W,   fc_bf,   EE * 2 * HH);
    cvt_f32_bf16_kernel<<<1024, 256, 0, stream>>>(emb,    emb_bf,  EE * EE);
    init_h_kernel<<<512, 256, 0, stream>>>(h_t, h_t_rev, h_state);
    init_c_kernel<<<512, 256, 0, stream>>>(c0, c0_rev, c_state);
    init_ended_kernel<<<1, 128, 0, stream>>>(ended);
    gather_first_kernel<<<BB, 256, 0, stream>>>(yy_pad, emb_bf, x_bf);

    // ---- sequential greedy decode ----
    for (int t = 0; t < TT; ++t) {
        const int pi = t & 1;        // ping (h in)
        const int po = (t + 1) & 1;  // pong (h out)

        // layer 0, fwd + rev (independent)
        lstm_cell_kernel<<<HH / 16, 256, 0, stream>>>(
            x_bf, hbuf(0, 0, pi), cbuf(0, 0),
            wih_bf, whh_bf, b_ih, b_hh, hbuf(0, 0, po));
        lstm_cell_kernel<<<HH / 16, 256, 0, stream>>>(
            x_bf, hbuf(0, 1, pi), cbuf(0, 1),
            wihr_bf, whhr_bf, b_ih_r, b_hh_r, hbuf(0, 1, po));

        // layer 1, fwd + rev (consume layer-0 outputs)
        lstm_cell_kernel<<<HH / 16, 256, 0, stream>>>(
            hbuf(0, 0, po), hbuf(1, 0, pi), cbuf(1, 0),
            wih_bf  + wsz / 2, whh_bf  + wsz / 2,
            b_ih + 4 * HH, b_hh + 4 * HH, hbuf(1, 0, po));
        lstm_cell_kernel<<<HH / 16, 256, 0, stream>>>(
            hbuf(0, 1, po), hbuf(1, 1, pi), cbuf(1, 1),
            wihr_bf + wsz / 2, whhr_bf + wsz / 2,
            b_ih_r + 4 * HH, b_hh_r + 4 * HH, hbuf(1, 1, po));

        // wh scramble, fc GEMM, argmax/softmax/embedding-gather
        repack_wh_kernel<<<(BB * 2 * HH + 255) / 256, 256, 0, stream>>>(
            hbuf(1, 0, po), hbuf(1, 1, po), wh_bf);
        fc_gemm_kernel<<<EE / 16, 256, 0, stream>>>(wh_bf, fc_bf, fc_b, logits);
        decide_kernel<<<BB, 256, 0, stream>>>(logits, ended, emb_bf, x_bf, out, t);
    }

    (void)in_sizes; (void)n_in; (void)out_size; (void)ws_size;
}
